// Attention_37366215475332
// MI455X (gfx1250) — matrix operand order repaired and checked by reading, not compiled, not run on hardware
//
#include <hip/hip_runtime.h>
#include <hip/hip_bf16.h>

// ---------------------------------------------------------------------------
// Problem constants (from reference)
// ---------------------------------------------------------------------------
#define BB   2
#define SS   2048
#define HID  2048
#define NH   32
#define HD   64
#define NQKV (HID + 2 * HD)   // 2176
#define MM   (BB * SS)        // 4096

typedef __attribute__((ext_vector_type(16))) __bf16 v16bf;
typedef __attribute__((ext_vector_type(8)))  __bf16 v8bf;
typedef __attribute__((ext_vector_type(8)))  float  v8f;

__device__ __forceinline__ unsigned short f32_to_bf16(float f) {
    unsigned int u = __float_as_uint(f);
    u += 0x7FFFu + ((u >> 16) & 1u);      // round-to-nearest-even
    return (unsigned short)(u >> 16);
}

// Load a v16bf fragment from two contiguous 8-element (16B) runs.
__device__ __forceinline__ v16bf ld_pair(const unsigned short* p0,
                                         const unsigned short* p1) {
    v8bf lo = *(const v8bf*)p0;
    v8bf hi = *(const v8bf*)p1;
    return __builtin_shufflevector(lo, hi, 0,1,2,3,4,5,6,7,8,9,10,11,12,13,14,15);
}

// CDNA5 async DMA: 16B global -> LDS, no VGPR data path, tracked by ASYNCcnt.
// Generic shared-memory pointers carry the LDS offset in their low 32 bits
// (ISA 10.2: LDS aperture addresses truncate to addr[31:0]).
__device__ __forceinline__ void async_b128(unsigned lds_byte, const void* gaddr) {
    asm volatile("global_load_async_to_lds_b128 %0, %1, off"
                 :: "v"(lds_byte), "v"(gaddr)
                 : "memory");
}

// ---------------------------------------------------------------------------
// Kernel 0a: plain f32 -> bf16 (RNE) conversion (A matrices)
// ---------------------------------------------------------------------------
__global__ __launch_bounds__(256) void cvt_bf16(const float* __restrict__ in,
                                                unsigned short* __restrict__ out,
                                                int n) {
    int i = blockIdx.x * blockDim.x + threadIdx.x;
    if (i < n) out[i] = f32_to_bf16(in[i]);
}

// ---------------------------------------------------------------------------
// Kernel 0b: f32 -> bf16 + pack weight [K,N] into WMMA-B fragment order:
//   out[((k>>5)*N + n)*32 + (k&31)]
// ---------------------------------------------------------------------------
__global__ __launch_bounds__(256) void wpack_bf16(const float* __restrict__ in,
                                                  unsigned short* __restrict__ out,
                                                  int K, int N) {
    int idx = blockIdx.x * blockDim.x + threadIdx.x;   // coalesced read index
    if (idx >= K * N) return;
    int k = idx / N;
    int n = idx - k * N;
    out[((size_t)(k >> 5) * N + n) * 32 + (k & 31)] = f32_to_bf16(in[idx]);
}

// ---------------------------------------------------------------------------
// Kernel 1/4: bf16 GEMM  C[M,N] = A[M,K] * Bp(packed)  (f32 accumulate)
//   WG = 128 threads = 4 waves; tile 128(M) x 64(N); k-step 32.
//   Double-buffered LDS fed by async-to-LDS DMA (6 ops/thread/slab).
// ---------------------------------------------------------------------------
__global__ __launch_bounds__(128) void gemm_bf16(const unsigned short* __restrict__ A,
                                                 const unsigned short* __restrict__ Bp,
                                                 float* __restrict__ C,
                                                 int M, int N, int K) {
    __shared__ unsigned short At[2][128 * 32];   // [m][k]         2 x 8 KB
    __shared__ unsigned short Bt[2][64 * 32];    // packed [n][k]  2 x 4 KB

    const int t    = threadIdx.x;
    const int wave = t >> 5;
    const int lane = t & 31;
    const int nloc = lane & 15;
    const int hh   = lane >> 4;
    const int m0   = blockIdx.y * 128;
    const int n0   = blockIdx.x * 64;

    v8f acc[2][4];
#pragma unroll
    for (int mt = 0; mt < 2; ++mt)
#pragma unroll
        for (int nt = 0; nt < 4; ++nt) acc[mt][nt] = v8f{};

    auto issueTile = [&](int k0, int buf) {
        // A: thread t owns row (m0+t): 32 bf16 = 4 x 16B
        const unsigned short* ga = A + (size_t)(m0 + t) * K + k0;
        unsigned la = (unsigned)(size_t)&At[buf][t * 32];
#pragma unroll
        for (int i = 0; i < 4; ++i) async_b128(la + 16 * i, ga + 8 * i);
        // B: packed slab region for [n0, n0+64) is contiguous (2048 elems)
        const unsigned short* gb = Bp + ((size_t)(k0 >> 5) * N + n0) * 32 + t * 16;
        unsigned lb = (unsigned)(size_t)&Bt[buf][t * 16];
        async_b128(lb, gb);
        async_b128(lb + 16, gb + 8);
    };

    issueTile(0, 0);
    int buf = 0;
    for (int k0 = 0; k0 < K; k0 += 32) {
        const bool hasNext = (k0 + 32) < K;        // block-uniform
        if (hasNext) {
            issueTile(k0 + 32, buf ^ 1);
            // <=6 outstanding == only the just-issued group may remain
            asm volatile("s_wait_asynccnt 0x6" ::: "memory");
        } else {
            asm volatile("s_wait_asynccnt 0x0" ::: "memory");
        }
        __syncthreads();

        // ---- fragments: everything is 2 x ds_load_b128 ----
        v16bf afrag[2];
#pragma unroll
        for (int mt = 0; mt < 2; ++mt) {
            const unsigned short* ap = At[buf] + (wave * 32 + mt * 16 + nloc) * 32;
            afrag[mt] = ld_pair(ap + 8 * hh, ap + 16 + 8 * hh);
        }
#pragma unroll
        for (int nt = 0; nt < 4; ++nt) {
            const unsigned short* bp = Bt[buf] + (nt * 16 + nloc) * 32;
            v16bf bfrag = ld_pair(bp + 8 * hh, bp + 16 + 8 * hh);
#pragma unroll
            for (int mt = 0; mt < 2; ++mt)
                acc[mt][nt] = __builtin_amdgcn_wmma_f32_16x16x32_bf16(
                    false, afrag[mt], false, bfrag, (short)0, acc[mt][nt], false, false);
        }
        __syncthreads();   // all waves done reading buf before it is re-filled
        buf ^= 1;
    }

    // Epilogue: C layout -> lane holds col n = lane&15, rows r + 8*hh
#pragma unroll
    for (int mt = 0; mt < 2; ++mt)
#pragma unroll
        for (int nt = 0; nt < 4; ++nt)
#pragma unroll
            for (int r = 0; r < 8; ++r) {
                int m = m0 + wave * 32 + mt * 16 + r + 8 * hh;
                int n = n0 + nt * 16 + nloc;
                C[(size_t)m * N + n] = acc[mt][nt][r];
            }
}

// ---------------------------------------------------------------------------
// Kernel 2: RoPE + repack  fused f32 [B,S,34,64] ->
//   q bf16 [B,NH,S,HD] (roped), k bf16 [B,S,HD] (roped),
//   v bf16 packed [B, S/32, HD, 32] (P*V B-fragment friendly)
// ---------------------------------------------------------------------------
__global__ __launch_bounds__(256) void rope_pack(const float* __restrict__ fused,
                                                 unsigned short* __restrict__ q,
                                                 unsigned short* __restrict__ k,
                                                 unsigned short* __restrict__ v) {
    int idx = blockIdx.x * blockDim.x + threadIdx.x;
    const int total = BB * SS * (NH + 2) * HD;
    if (idx >= total) return;
    int d = idx & (HD - 1);
    int j = (idx >> 6) % (NH + 2);
    int s = (idx >> 6) / (NH + 2) % SS;
    int b = idx / (HD * (NH + 2) * SS);

    const float* row = fused + ((size_t)(b * SS + s) * (NH + 2) + j) * HD;
    float val = row[d];
    float outv;
    if (j <= NH) {  // q heads (j<32) and k (j==32) get RoPE
        int   i    = d & 31;
        float invf = __powf(10000.0f, -((float)i) / 32.0f);
        float ang  = (float)s * invf;
        float c = __cosf(ang), sn = __sinf(ang);
        float partner = (d < 32) ? -row[d + 32] : row[d - 32];
        outv = val * c + partner * sn;
    } else {
        outv = val;  // v passthrough
    }
    unsigned short o = f32_to_bf16(outv);
    if (j < NH)
        q[((size_t)(b * NH + j) * SS + s) * HD + d] = o;
    else if (j == NH)
        k[((size_t)(b * SS) + s) * HD + d] = o;
    else  // packed V: [b][s>>5][d][s&31]
        v[((size_t)(b * (SS / 32) + (s >> 5)) * HD + d) * 32 + (s & 31)] = o;
}

// ---------------------------------------------------------------------------
// Kernel 3: causal flash attention (MQA: 1 KV head shared by 32 Q heads)
//   grid = (S/64, NH, B), block = 128 (4 waves x 16 queries)
//   Double-buffered async K/V staging; 8 WMMA per 32-key block per wave.
// ---------------------------------------------------------------------------
__global__ __launch_bounds__(128) void flash_attn(const unsigned short* __restrict__ Q,
                                                  const unsigned short* __restrict__ Kc,
                                                  const unsigned short* __restrict__ Vp,
                                                  unsigned short* __restrict__ Out) {
    __shared__ unsigned short Klds[2][32 * 64];   // [key][hd]       2 x 4 KB
    __shared__ unsigned short Vlds[2][64 * 32];   // packed [hd][key] 2 x 4 KB
    __shared__ unsigned short Plds[4][16 * 32];   // per-wave P tile [m][key]

    const int b     = blockIdx.z;
    const int h     = blockIdx.y;
    const int qtile = blockIdx.x * 64;
    const int t     = threadIdx.x;
    const int wave  = t >> 5;
    const int lane  = t & 31;
    const int nloc  = lane & 15;
    const int hh    = lane >> 4;
    const int q0    = qtile + wave * 16;

    // Q fragments (A layout, row = q0 + lane&15), loaded once from global
    const unsigned short* qbase =
        Q + ((size_t)(b * NH + h) * SS + q0 + nloc) * HD;
    v16bf aq0 = ld_pair(qbase + 8 * hh,      qbase + 16 + 8 * hh);
    v16bf aq1 = ld_pair(qbase + 32 + 8 * hh, qbase + 48 + 8 * hh);

    float mrow[8], lrow[8];
    v8f   o[4];
#pragma unroll
    for (int r = 0; r < 8; ++r) { mrow[r] = -1e30f; lrow[r] = 0.0f; }
#pragma unroll
    for (int vt = 0; vt < 4; ++vt) o[vt] = v8f{};

    const float scale = 0.125f;  // 1/sqrt(64)
    const int rowL = t >> 2, colL = (t & 3) * 16;

    auto issueKV = [&](int kb, int buf) {
        const unsigned short* gk = Kc + ((size_t)(b * SS + kb + rowL)) * HD + colL;
        unsigned lk = (unsigned)(size_t)&Klds[buf][rowL * 64 + colL];
        async_b128(lk, gk);
        async_b128(lk + 16, gk + 8);
        const unsigned short* gv =
            Vp + (size_t)(b * (SS / 32) + (kb >> 5)) * HD * 32 + t * 16;
        unsigned lv = (unsigned)(size_t)&Vlds[buf][t * 16];
        async_b128(lv, gv);
        async_b128(lv + 16, gv + 8);
    };

    issueKV(0, 0);
    int buf = 0;
    for (int kb = 0; kb < qtile + 64; kb += 32) {
        const bool hasNext = (kb + 32) < (qtile + 64);   // block-uniform
        if (hasNext) {
            issueKV(kb + 32, buf ^ 1);
            asm volatile("s_wait_asynccnt 0x4" ::: "memory");
        } else {
            asm volatile("s_wait_asynccnt 0x0" ::: "memory");
        }
        __syncthreads();

        if (kb <= q0 + 15) {   // wave-uniform: EXEC stays all-1s for WMMA
            // ---- scores: two 16x16 tiles over the 32-key block ----
            v8f sacc[2];
#pragma unroll
            for (int nt = 0; nt < 2; ++nt) {
                const unsigned short* kp = Klds[buf] + (nt * 16 + nloc) * 64;  // col = key
                v16bf b0 = ld_pair(kp + 8 * hh,      kp + 16 + 8 * hh);       // hd 0..31
                v16bf b1 = ld_pair(kp + 32 + 8 * hh, kp + 48 + 8 * hh);  // hd 32..63
                v8f z = {};
                v8f s_ = __builtin_amdgcn_wmma_f32_16x16x32_bf16(
                    false, aq0, false, b0, (short)0, z, false, false);
                s_ = __builtin_amdgcn_wmma_f32_16x16x32_bf16(
                    false, aq1, false, b1, (short)0, s_, false, false);
                sacc[nt] = s_;
            }

            // ---- online softmax over the 32 keys (rows r + 8*hh) ----
#pragma unroll
            for (int r = 0; r < 8; ++r) {
                int qg = q0 + r + 8 * hh;
                float s0v = sacc[0][r] * scale;
                if (kb + nloc > qg) s0v = -1e30f;
                float s1v = sacc[1][r] * scale;
                if (kb + 16 + nloc > qg) s1v = -1e30f;

                float tmax = fmaxf(s0v, s1v);
#pragma unroll
                for (int off = 1; off < 16; off <<= 1)
                    tmax = fmaxf(tmax, __shfl_xor(tmax, off, 32));
                float mnew = fmaxf(mrow[r], tmax);

                float e0 = __expf(s0v - mnew);
                float e1 = __expf(s1v - mnew);
                float rs = e0 + e1;
#pragma unroll
                for (int off = 1; off < 16; off <<= 1)
                    rs += __shfl_xor(rs, off, 32);

                float corr = __expf(mrow[r] - mnew);
                lrow[r] = lrow[r] * corr + rs;
                mrow[r] = mnew;
#pragma unroll
                for (int vt = 0; vt < 4; ++vt) o[vt][r] *= corr;

                int m = r + 8 * hh;
                Plds[wave][m * 32 + nloc]      = f32_to_bf16(e0);
                Plds[wave][m * 32 + 16 + nloc] = f32_to_bf16(e1);
            }

            // wave-internal DS RAW: P tile written above, read below
            asm volatile("s_wait_dscnt 0x0" ::: "memory");

            // ---- P * V: A = P [16 x 32 keys], B = packed V ----
            const unsigned short* pp = &Plds[wave][0];
            v16bf ap = ld_pair(pp + nloc * 32 + 8 * hh,
                               pp + nloc * 32 + 16 + 8 * hh);
#pragma unroll
            for (int vt = 0; vt < 4; ++vt) {
                const unsigned short* vb = Vlds[buf] + (vt * 16 + nloc) * 32;
                v16bf bv = ld_pair(vb + 8 * hh, vb + 16 + 8 * hh);
                o[vt] = __builtin_amdgcn_wmma_f32_16x16x32_bf16(
                    false, ap, false, bv, (short)0, o[vt], false, false);
            }
        }
        __syncthreads();   // buffer reuse fence
        buf ^= 1;
    }

    // epilogue: normalize and write attn_out bf16 [B,S,NH*HD]
#pragma unroll
    for (int r = 0; r < 8; ++r) {
        float inv = 1.0f / lrow[r];
        int qg = q0 + r + 8 * hh;
        size_t rowoff = (size_t)(b * SS + qg) * (NH * HD) + h * HD;
#pragma unroll
        for (int vt = 0; vt < 4; ++vt)
            Out[rowoff + vt * 16 + nloc] = f32_to_bf16(o[vt][r] * inv);
    }
}

// ---------------------------------------------------------------------------
// Host launcher
// ---------------------------------------------------------------------------
extern "C" void kernel_launch(void* const* d_in, const int* in_sizes, int n_in,
                              void* d_out, int out_size, void* d_ws, size_t ws_size,
                              hipStream_t stream) {
    const float* hidden  = (const float*)d_in[0];
    const float* w_qkv   = (const float*)d_in[1];
    const float* w_dense = (const float*)d_in[2];
    float* out = (float*)d_out;

    char* ws = (char*)d_ws;
    size_t off = 0;
    auto carve = [&](size_t bytes) -> char* {
        char* p = ws + off;
        off += (bytes + 255) & ~(size_t)255;
        return p;
    };

    unsigned short* hid_bf  = (unsigned short*)carve((size_t)MM * HID * 2);
    unsigned short* wqkv_bf = (unsigned short*)carve((size_t)HID * NQKV * 2);
    unsigned short* wd_bf   = (unsigned short*)carve((size_t)HID * HID * 2);
    float*          fused   = (float*)carve((size_t)MM * NQKV * 4);
    unsigned short* q_bf    = (unsigned short*)carve((size_t)BB * NH * SS * HD * 2);
    unsigned short* k_bf    = (unsigned short*)carve((size_t)BB * SS * HD * 2);
    unsigned short* v_bf    = (unsigned short*)carve((size_t)BB * SS * HD * 2);
    unsigned short* attn_bf = (unsigned short*)carve((size_t)MM * HID * 2);

    // 1) conversions: A matrices plain, weights packed into B-fragment order
    {
        int n = MM * HID;
        cvt_bf16<<<(n + 255) / 256, 256, 0, stream>>>(hidden, hid_bf, n);
        n = HID * NQKV;
        wpack_bf16<<<(n + 255) / 256, 256, 0, stream>>>(w_qkv, wqkv_bf, HID, NQKV);
        n = HID * HID;
        wpack_bf16<<<(n + 255) / 256, 256, 0, stream>>>(w_dense, wd_bf, HID, HID);
    }

    // 2) QKV GEMM: [4096 x 2048] * [2048 x 2176] -> fused f32
    gemm_bf16<<<dim3(NQKV / 64, MM / 128), 128, 0, stream>>>(
        hid_bf, wqkv_bf, fused, MM, NQKV, HID);

    // 3) RoPE + repack into q/k/v bf16 (V packed for P*V fragments)
    {
        int n = BB * SS * (NH + 2) * HD;
        rope_pack<<<(n + 255) / 256, 256, 0, stream>>>(fused, q_bf, k_bf, v_bf);
    }

    // 4) causal flash attention -> attn_bf [4096 x 2048]
    flash_attn<<<dim3(SS / 64, NH, BB), 128, 0, stream>>>(q_bf, k_bf, v_bf, attn_bf);

    // 5) dense projection: [4096 x 2048] * [2048 x 2048] -> d_out f32
    gemm_bf16<<<dim3(HID / 64, MM / 128), 128, 0, stream>>>(
        attn_bf, wd_bf, out, MM, HID, HID);
}
